// RWKV7_TimeMix_88708254532083
// MI455X (gfx1250) — matrix operand order repaired and checked by reading, not compiled, not run on hardware
//
#include <hip/hip_runtime.h>

// ---------------------------------------------------------------------------
// Types for CDNA5 WMMA
// ---------------------------------------------------------------------------
typedef __attribute__((ext_vector_type(16))) __bf16 v16bf;
typedef __attribute__((ext_vector_type(8)))  float  v8f;

struct frag32 { uint4 lo; uint4 hi; };   // 32 bytes = 16 bf16

__device__ __forceinline__ unsigned short f2bf(float f) {
    unsigned int u = __float_as_uint(f);
    u += 0x7FFFu + ((u >> 16) & 1u);   // round-to-nearest-even
    return (unsigned short)(u >> 16);
}

// ---------------------------------------------------------------------------
// Time-shift + six token mixes -> bf16 activations
// ---------------------------------------------------------------------------
__global__ void mix_kernel(const float* __restrict__ x,
                           const float* __restrict__ mr, const float* __restrict__ mw,
                           const float* __restrict__ mk, const float* __restrict__ mv,
                           const float* __restrict__ ma, const float* __restrict__ mg,
                           unsigned short* __restrict__ xr, unsigned short* __restrict__ xw,
                           unsigned short* __restrict__ xk, unsigned short* __restrict__ xv,
                           unsigned short* __restrict__ xa, unsigned short* __restrict__ xg,
                           int T, int C, long total)
{
    long i = (long)blockIdx.x * blockDim.x + threadIdx.x;
    if (i >= total) return;
    int  c   = (int)(i % C);
    long row = i / C;
    int  t   = (int)(row % T);
    float xi   = x[i];
    float prev = (t > 0) ? x[i - C] : 0.0f;
    float xx   = prev - xi;
    xr[i] = f2bf(xi + xx * mr[c]);
    xw[i] = f2bf(xi + xx * mw[c]);
    xk[i] = f2bf(xi + xx * mk[c]);
    xv[i] = f2bf(xi + xx * mv[c]);
    xa[i] = f2bf(xi + xx * ma[c]);
    xg[i] = f2bf(xi + xx * mg[c]);
}

// ---------------------------------------------------------------------------
// Elementwise conversions / activations
// ---------------------------------------------------------------------------
__global__ void cvt_f32_bf16(const float* __restrict__ in, unsigned short* __restrict__ out, long n) {
    long i = (long)blockIdx.x * blockDim.x + threadIdx.x;
    if (i < n) out[i] = f2bf(in[i]);
}
// Weight convert + transpose: in[K][N] f32 -> out[N][K] bf16 (B operand layout)
__global__ void cvt_transpose_bf16(const float* __restrict__ in, unsigned short* __restrict__ out,
                                   int K, int N) {
    long i = (long)blockIdx.x * blockDim.x + threadIdx.x;
    if (i >= (long)K * N) return;
    int k = (int)(i / N);
    int n = (int)(i % N);
    out[(size_t)n * K + k] = f2bf(in[i]);
}
__global__ void tanh_bf16(const float* __restrict__ in, unsigned short* __restrict__ out, long n) {
    long i = (long)blockIdx.x * blockDim.x + threadIdx.x;
    if (i < n) out[i] = f2bf(tanhf(in[i]));
}
__global__ void sigmoid_bf16(const float* __restrict__ in, unsigned short* __restrict__ out, long n) {
    long i = (long)blockIdx.x * blockDim.x + threadIdx.x;
    if (i < n) out[i] = f2bf(1.0f / (1.0f + expf(-in[i])));
}

// ---------------------------------------------------------------------------
// bf16 GEMM via V_WMMA_F32_16X16X32_BF16.
// C[M,N] = A[M,K] @ B[K,N];  A bf16 row-major [M][K], B pre-transposed [N][K].
// Block: 256 threads (8 wave32), 128x64 output tile, K stepped by 32.
// Wave w computes a 32x32 output patch (2x2 subtiles, 4 WMMAs) at
//   rows m0 + (w>>1)*32 + {0,16},  cols n0 + (w&1)*32 + {0,16}.
// All fragments are contiguous 16B runs in LDS -> ds_load_b128 only;
// A/B fragments each reused twice from registers (4 WMMA : 8 ds_load_b128).
// ---------------------------------------------------------------------------
template<int N, int K>
__global__ __launch_bounds__(256) void gemm_bf16(const unsigned short* __restrict__ A,
                                                 const unsigned short* __restrict__ Bt,
                                                 float* __restrict__ Cmat, int M)
{
    __shared__ unsigned short As[128][32];  // [m][k] tile (8 KB)
    __shared__ unsigned short Bs[64][32];   // [n][k] tile, transposed B (4 KB)

    const int tid  = threadIdx.x;
    const int wave = tid >> 5;
    const int lane = tid & 31;
    const int m0   = blockIdx.y * 128;
    const int n0   = blockIdx.x * 64;

    // staging coordinates
    const int raA = tid >> 1;            // 0..127
    const int caA = (tid & 1) * 16;      // 0 or 16 (32B run)
    const int raB = tid >> 2;            // 0..63
    const int caB = (tid & 3) * 8;       // 0,8,16,24

    const unsigned short* pA = A  + (size_t)(m0 + raA) * K + caA;
    const unsigned short* pB = Bt + (size_t)(n0 + raB) * K + caB;

    // fragment coordinates
    const int mi   = (wave >> 1) * 32;   // 0,32,64,96
    const int nj   = (wave & 1) * 32;    // 0,32
    const int lrow = lane & 15;
    const int koff = (lane >> 4) * 8;    // A half-wave K offset
    const int kb   = (lane >> 4) * 8;    // B half-wave K offset
    const int mr0  = mi + lrow, mr1 = mi + 16 + lrow;
    const int nc0  = nj + lrow, nc1 = nj + 16 + lrow;

    v8f acc00 = {}, acc01 = {}, acc10 = {}, acc11 = {};

    #pragma unroll 2
    for (int k0 = 0; k0 < K; k0 += 32) {
        *(uint4*)(&As[raA][caA])     = *(const uint4*)(pA + k0);
        *(uint4*)(&As[raA][caA + 8]) = *(const uint4*)(pA + k0 + 8);
        *(uint4*)(&Bs[raB][caB])     = *(const uint4*)(pB + k0);
        if (k0 + 32 < K) {   // next tiles into cache (global_prefetch_b8)
            __builtin_prefetch(pA + k0 + 32, 0, 1);
            __builtin_prefetch(pB + k0 + 32, 0, 1);
        }
        __syncthreads();

        frag32 fa0, fa1, fb0, fb1;
        fa0.lo = *(const uint4*)(&As[mr0][koff]);
        fa0.hi = *(const uint4*)(&As[mr0][16 + koff]);
        fa1.lo = *(const uint4*)(&As[mr1][koff]);
        fa1.hi = *(const uint4*)(&As[mr1][16 + koff]);
        fb0.lo = *(const uint4*)(&Bs[nc0][kb]);
        fb0.hi = *(const uint4*)(&Bs[nc0][kb + 16]);
        fb1.lo = *(const uint4*)(&Bs[nc1][kb]);
        fb1.hi = *(const uint4*)(&Bs[nc1][kb + 16]);

        v16bf a0 = __builtin_bit_cast(v16bf, fa0);
        v16bf a1 = __builtin_bit_cast(v16bf, fa1);
        v16bf b0 = __builtin_bit_cast(v16bf, fb0);
        v16bf b1 = __builtin_bit_cast(v16bf, fb1);

        acc00 = __builtin_amdgcn_wmma_f32_16x16x32_bf16(false, a0, false, b0, (short)0, acc00, false, false);
        acc01 = __builtin_amdgcn_wmma_f32_16x16x32_bf16(false, a0, false, b1, (short)0, acc01, false, false);
        acc10 = __builtin_amdgcn_wmma_f32_16x16x32_bf16(false, a1, false, b0, (short)0, acc10, false, false);
        acc11 = __builtin_amdgcn_wmma_f32_16x16x32_bf16(false, a1, false, b1, (short)0, acc11, false, false);
        __syncthreads();
    }

    // C store: VGPR i holds row (i + 8*(lane/16)); N compile-time -> immediates
    {
        const int mb0 = m0 + mi + (lane >> 4) * 8;        // subtile row block 0
        const int mb1 = mb0 + 16;                          // subtile row block 1
        float* cp0 = Cmat + (size_t)mb0 * N + n0;
        float* cp1 = Cmat + (size_t)mb1 * N + n0;
        #pragma unroll
        for (int i = 0; i < 8; ++i) {
            cp0[(size_t)i * N + nc0] = acc00[i];
            cp0[(size_t)i * N + nc1] = acc01[i];
            cp1[(size_t)i * N + nc0] = acc10[i];
            cp1[(size_t)i * N + nc1] = acc11[i];
        }
    }
}

// ---------------------------------------------------------------------------
// Pre-scan fusion: decay w, sigmoid a, L2-normalized removal key, k rescale.
// One block = one (row, head); 64 threads over N. In-place aliasing:
// wdec<-wlora, aa<-alora, knew<-k (each thread reads its element first).
// ---------------------------------------------------------------------------
__global__ __launch_bounds__(64) void prescan_kernel(const float* k, const float* wlora,
                                                     const float* alora,
                                                     const float* __restrict__ w0,
                                                     const float* __restrict__ a0,
                                                     const float* __restrict__ k_k,
                                                     const float* __restrict__ k_a,
                                                     float* wdec, float* knew,
                                                     float* aa, float* __restrict__ bb, int H)
{
    __shared__ float red[64];
    int  n   = threadIdx.x;
    long row = blockIdx.x / H;
    int  h   = blockIdx.x % H;
    int  c   = h * 64 + n;
    long i   = row * (long)(H * 64) + c;

    float kf  = k[i];
    float wl  = wlora[i];
    float al  = alora[i];
    float kkv = kf * k_k[c];

    red[n] = kkv * kkv;
    __syncthreads();
    #pragma unroll
    for (int s = 32; s > 0; s >>= 1) { if (n < s) red[n] += red[n + s]; __syncthreads(); }
    float norm = sqrtf(red[0]);
    float kkn  = kkv / fmaxf(norm, 1e-12f);

    float av = 1.0f / (1.0f + expf(-(a0[c] + al)));
    float z  = w0[c] + wl;
    float wv = -log1pf(expf(-z)) - 0.5f;      // -softplus(-z) - 0.5

    wdec[i] = expf(-expf(wv));
    aa[i]   = -kkn;
    bb[i]   = kkn * av;
    knew[i] = kf * (1.0f + (av - 1.0f) * k_a[c]);
}

// ---------------------------------------------------------------------------
// Serial delta-rule scan. One block per (b,h); 64 threads; thread i owns
// state row i (64 f32 in registers). Per-step vectors staged in LDS.
// ---------------------------------------------------------------------------
__global__ __launch_bounds__(64) void scan_kernel(const float* __restrict__ r,
                                                  const float* __restrict__ wdec,
                                                  const float* __restrict__ knew,
                                                  const float* __restrict__ v,
                                                  const float* __restrict__ aa,
                                                  const float* __restrict__ bb,
                                                  float* __restrict__ y,
                                                  int Bsz, int T, int H)
{
    int bh = blockIdx.x;
    int b  = bh / H;
    int h  = bh % H;
    int n  = threadIdx.x;
    const int C = H * 64;

    __shared__ float rv[64], wv[64], kv[64], av[64], bv[64];
    float st[64];
    #pragma unroll
    for (int j = 0; j < 64; ++j) st[j] = 0.0f;

    for (int t = 0; t < T; ++t) {
        long base = ((long)(b * T + t)) * C + h * 64;
        rv[n] = r[base + n];
        wv[n] = wdec[base + n];
        kv[n] = knew[base + n];
        av[n] = aa[base + n];
        bv[n] = bb[base + n];
        float vi = v[base + n];
        __syncthreads();

        float sa = 0.0f;
        #pragma unroll
        for (int j = 0; j < 64; ++j) sa += st[j] * av[j];

        float o = 0.0f;
        #pragma unroll
        for (int j = 0; j < 64; ++j) {
            st[j] = st[j] * wv[j] + sa * bv[j] + vi * kv[j];
            o += st[j] * rv[j];
        }
        y[base + n] = o;
        __syncthreads();
    }
}

// ---------------------------------------------------------------------------
// GroupNorm (per head, eps=6.4e-4) + bonus term + gate -> bf16
// ---------------------------------------------------------------------------
__device__ __forceinline__ float reduce64(float* red, int n, float val) {
    __syncthreads();
    red[n] = val;
    __syncthreads();
    #pragma unroll
    for (int s = 32; s > 0; s >>= 1) { if (n < s) red[n] += red[n + s]; __syncthreads(); }
    return red[0];
}

__global__ __launch_bounds__(64) void postnorm_kernel(const float* __restrict__ y,
                                                      const float* __restrict__ r,
                                                      const float* __restrict__ knew,
                                                      const float* __restrict__ v,
                                                      const float* __restrict__ g,
                                                      const float* __restrict__ r_k,
                                                      const float* __restrict__ gamma,
                                                      const float* __restrict__ beta,
                                                      unsigned short* __restrict__ ygated,
                                                      int H, float eps)
{
    __shared__ float red[64];
    int  n   = threadIdx.x;
    long row = blockIdx.x / H;
    int  h   = blockIdx.x % H;
    int  c   = h * 64 + n;
    long i   = row * (long)(H * 64) + c;

    float yv = y[i];
    float mu = reduce64(red, n, yv) * (1.0f / 64.0f);
    float d  = yv - mu;
    float var = reduce64(red, n, d * d) * (1.0f / 64.0f);
    float bonus = reduce64(red, n, r[i] * knew[i] * r_k[c]);

    float yn = d * rsqrtf(var + eps) * gamma[c] + beta[c];
    float yf = yn + bonus * v[i];
    ygated[i] = f2bf(yf * g[i]);
}

// ---------------------------------------------------------------------------
// Host-side launch
// ---------------------------------------------------------------------------
template<int N, int K>
static inline void launch_gemm(const unsigned short* A, const unsigned short* Bt,
                               float* Cm, int M, hipStream_t s)
{
    gemm_bf16<N, K><<<dim3((unsigned)(N / 64), (unsigned)(M / 128)), 256, 0, s>>>(A, Bt, Cm, M);
}

extern "C" void kernel_launch(void* const* d_in, const int* in_sizes, int n_in,
                              void* d_out, int out_size, void* d_ws, size_t ws_size,
                              hipStream_t stream)
{
    (void)in_sizes; (void)n_in; (void)out_size; (void)ws_size;
    const int Bsz = 4, T = 1024, C = 768, H = 12;
    const int ME  = Bsz * T;              // 4096
    const long MC = (long)ME * C;

    const float* x    = (const float*)d_in[0];
    const float* x_r  = (const float*)d_in[1];
    const float* x_w  = (const float*)d_in[2];
    const float* x_k  = (const float*)d_in[3];
    const float* x_v  = (const float*)d_in[4];
    const float* x_a  = (const float*)d_in[5];
    const float* x_g  = (const float*)d_in[6];
    const float* w0   = (const float*)d_in[7];
    const float* w1   = (const float*)d_in[8];
    const float* w2   = (const float*)d_in[9];
    const float* a0   = (const float*)d_in[10];
    const float* a1   = (const float*)d_in[11];
    const float* a2   = (const float*)d_in[12];
    const float* g1   = (const float*)d_in[13];
    const float* g2   = (const float*)d_in[14];
    const float* k_k  = (const float*)d_in[15];
    const float* k_a  = (const float*)d_in[16];
    const float* r_k  = (const float*)d_in[17];
    const float* Wr   = (const float*)d_in[18];
    const float* Wk   = (const float*)d_in[19];
    const float* Wv   = (const float*)d_in[20];
    const float* Wo   = (const float*)d_in[21];
    const float* gam  = (const float*)d_in[22];
    const float* bet  = (const float*)d_in[23];

    char*  ws  = (char*)d_ws;
    size_t off = 0;
    auto alloc = [&](size_t bytes) -> void* {
        void* p = ws + off;
        off += (bytes + 255) & ~(size_t)255;
        return p;
    };

    // bf16 activations (mixes)
    unsigned short* xr_b = (unsigned short*)alloc(MC * 2);
    unsigned short* xw_b = (unsigned short*)alloc(MC * 2);
    unsigned short* xk_b = (unsigned short*)alloc(MC * 2);
    unsigned short* xv_b = (unsigned short*)alloc(MC * 2);
    unsigned short* xa_b = (unsigned short*)alloc(MC * 2);
    unsigned short* xg_b = (unsigned short*)alloc(MC * 2);
    // bf16 weights, pre-transposed to [N][K]
    unsigned short* wr_b = (unsigned short*)alloc((size_t)C * C * 2);
    unsigned short* wk_b = (unsigned short*)alloc((size_t)C * C * 2);
    unsigned short* wv_b = (unsigned short*)alloc((size_t)C * C * 2);
    unsigned short* wo_b = (unsigned short*)alloc((size_t)C * C * 2);
    unsigned short* w1_b = (unsigned short*)alloc((size_t)C * 64 * 2);
    unsigned short* w2_b = (unsigned short*)alloc((size_t)64 * C * 2);
    unsigned short* a1_b = (unsigned short*)alloc((size_t)C * 64 * 2);
    unsigned short* a2_b = (unsigned short*)alloc((size_t)64 * C * 2);
    unsigned short* g1_b = (unsigned short*)alloc((size_t)C * 128 * 2);
    unsigned short* g2_b = (unsigned short*)alloc((size_t)128 * C * 2);
    // f32 large buffers
    float* r_f  = (float*)alloc(MC * 4);
    float* k_f  = (float*)alloc(MC * 4);   // -> knew (in place)
    float* v_f  = (float*)alloc(MC * 4);
    float* wl_f = (float*)alloc(MC * 4);   // -> wdec (in place)
    float* al_f = (float*)alloc(MC * 4);   // -> aa (in place)
    float* g_f  = (float*)alloc(MC * 4);
    float* bb_f = (float*)alloc(MC * 4);
    float* y_f  = (float*)alloc(MC * 4);
    // LoRA intermediates
    float*          hw_f = (float*)alloc((size_t)ME * 64 * 4);
    float*          ha_f = (float*)alloc((size_t)ME * 64 * 4);
    float*          hg_f = (float*)alloc((size_t)ME * 128 * 4);
    unsigned short* hw_b = (unsigned short*)alloc((size_t)ME * 64 * 2);
    unsigned short* ha_b = (unsigned short*)alloc((size_t)ME * 64 * 2);
    unsigned short* hg_b = (unsigned short*)alloc((size_t)ME * 128 * 2);
    unsigned short* yg_b = (unsigned short*)alloc(MC * 2);

    // 1) time shift + mixes
    mix_kernel<<<dim3((unsigned)((MC + 255) / 256)), 256, 0, stream>>>(
        x, x_r, x_w, x_k, x_v, x_a, x_g,
        xr_b, xw_b, xk_b, xv_b, xa_b, xg_b, T, C, MC);

    // 2) weights -> bf16, transposed to [N][K]
    auto cvtT = [&](const float* src, unsigned short* dst, int K, int N) {
        long n = (long)K * N;
        cvt_transpose_bf16<<<dim3((unsigned)((n + 255) / 256)), 256, 0, stream>>>(src, dst, K, N);
    };
    cvtT(Wr, wr_b, C, C);   cvtT(Wk, wk_b, C, C);
    cvtT(Wv, wv_b, C, C);   cvtT(Wo, wo_b, C, C);
    cvtT(w1, w1_b, C, 64);  cvtT(w2, w2_b, 64, C);
    cvtT(a1, a1_b, C, 64);  cvtT(a2, a2_b, 64, C);
    cvtT(g1, g1_b, C, 128); cvtT(g2, g2_b, 128, C);

    // 3) WMMA GEMMs (projections + LoRA down)
    launch_gemm<768, 768>(xr_b, wr_b, r_f,  ME, stream);
    launch_gemm<768, 768>(xk_b, wk_b, k_f,  ME, stream);
    launch_gemm<768, 768>(xv_b, wv_b, v_f,  ME, stream);
    launch_gemm< 64, 768>(xw_b, w1_b, hw_f, ME, stream);
    launch_gemm< 64, 768>(xa_b, a1_b, ha_f, ME, stream);
    launch_gemm<128, 768>(xg_b, g1_b, hg_f, ME, stream);

    // 4) activations between LoRA stages
    tanh_bf16<<<dim3((unsigned)(((long)ME * 64 + 255) / 256)), 256, 0, stream>>>(hw_f, hw_b, (long)ME * 64);
    cvt_f32_bf16<<<dim3((unsigned)(((long)ME * 64 + 255) / 256)), 256, 0, stream>>>(ha_f, ha_b, (long)ME * 64);
    sigmoid_bf16<<<dim3((unsigned)(((long)ME * 128 + 255) / 256)), 256, 0, stream>>>(hg_f, hg_b, (long)ME * 128);

    // 5) LoRA up-projections
    launch_gemm<768,  64>(hw_b, w2_b, wl_f, ME, stream);
    launch_gemm<768,  64>(ha_b, a2_b, al_f, ME, stream);
    launch_gemm<768, 128>(hg_b, g2_b, g_f,  ME, stream);

    // 6) pre-scan fusion (in-place: wl_f->wdec, al_f->aa, k_f->knew)
    prescan_kernel<<<dim3((unsigned)(ME * H)), 64, 0, stream>>>(
        k_f, wl_f, al_f, w0, a0, k_k, k_a, wl_f, k_f, al_f, bb_f, H);

    // 7) serial per-head delta-rule scan
    scan_kernel<<<dim3((unsigned)(Bsz * H)), 64, 0, stream>>>(
        r_f, wl_f, k_f, v_f, al_f, bb_f, y_f, Bsz, T, H);

    // 8) GroupNorm + bonus + gate
    postnorm_kernel<<<dim3((unsigned)(ME * H)), 64, 0, stream>>>(
        y_f, r_f, k_f, v_f, g_f, r_k, gam, bet, yg_b, H, 0.00064f);

    // 9) output projection -> d_out (f32)
    launch_gemm<768, 768>(yg_b, wo_b, (float*)d_out, ME, stream);
}
